// EncoderLayer_78288663871917
// MI455X (gfx1250) — compile-verified, matrix operand order checked by reading, not run on hardware
//
#include <hip/hip_runtime.h>
#include <hip/hip_bf16.h>
#include <math.h>

// ---------------------------------------------------------------------------
// Transformer encoder layer for MI455X (gfx1250, wave32, WMMA + TDM).
// GEMMs: v_wmma_f32_16x16x32_bf16 with TDM (tensor_load_to_lds) double-buffered
// LDS staging. f32 kept for softmax / layernorm / residuals / attn output.
// ---------------------------------------------------------------------------

typedef __bf16 bf16;
typedef __attribute__((ext_vector_type(16))) __bf16    v16bf;
typedef __attribute__((ext_vector_type(8)))  float     v8f;
typedef __attribute__((ext_vector_type(4)))  float     v4f;
typedef __attribute__((ext_vector_type(4)))  unsigned  v4u;
typedef __attribute__((ext_vector_type(8)))  unsigned  v8u;

#define D_MODEL 1024
#define N_HEADS 16
#define D_FF    4096
#define BATCH   2
#define SEQ     2048
#define NTOK    (BATCH*SEQ)      // 4096
#define DK      64
#define LN_EPS  1e-6f

enum { MODE_QK = 0, MODE_V = 1, MODE_SCORES = 2, MODE_RELU = 3, MODE_RESID = 4, MODE_CTX = 5 };

union FragU { uint4 u4[2]; v16bf v; };

// bf16 fragment per documented 16-bit WMMA A/B layout: lane l, hi = lane>=16;
// elems 0..7 -> K = hi*8 + i ; elems 8..15 -> K = 16 + hi*8 + i (two 16B loads)
__device__ __forceinline__ v16bf load_frag(const bf16* __restrict__ row, int kb, int hi) {
    FragU f;
    f.u4[0] = *(const uint4*)(row + kb + hi * 8);
    f.u4[1] = *(const uint4*)(row + kb + 16 + hi * 8);
    return f.v;
}

// Same fragment from f32 source (attention probabilities), nontemporal.
__device__ __forceinline__ v16bf load_frag_f32(const float* __restrict__ row, int kb, int hi) {
    const v4f* p0 = (const v4f*)(row + kb + hi * 8);
    const v4f* p1 = (const v4f*)(row + kb + 16 + hi * 8);
    v4f a = __builtin_nontemporal_load(p0);
    v4f b = __builtin_nontemporal_load(p0 + 1);
    v4f c = __builtin_nontemporal_load(p1);
    v4f d = __builtin_nontemporal_load(p1 + 1);
    v16bf r;
    r[0]  = (bf16)a[0]; r[1]  = (bf16)a[1]; r[2]  = (bf16)a[2]; r[3]  = (bf16)a[3];
    r[4]  = (bf16)b[0]; r[5]  = (bf16)b[1]; r[6]  = (bf16)b[2]; r[7]  = (bf16)b[3];
    r[8]  = (bf16)c[0]; r[9]  = (bf16)c[1]; r[10] = (bf16)c[2]; r[11] = (bf16)c[3];
    r[12] = (bf16)d[0]; r[13] = (bf16)d[1]; r[14] = (bf16)d[2]; r[15] = (bf16)d[3];
    return r;
}

__device__ __forceinline__ v8f wmma_bf16(v16bf a, v16bf b, v8f c) {
    // (neg_a, A, neg_b, B, c_mod, C, reuse_a, reuse_b)
    return __builtin_amdgcn_wmma_f32_16x16x32_bf16(false, a, false, b, (short)0, c, false, false);
}

// ---- Tensor Data Mover: 2D tile load Global -> LDS  (ISA 08_async_tensor §8)
// D# group0: [1:0]=count=1, [63:32]=lds_addr, [120:64]=global_addr, [127:126]=type(2)
// D# group1: [17:16]=data_size(1 => 2B), [79:48]=tensor_dim0, [111:80]=tensor_dim1,
//            [127:112]=tile_dim0, [143:128]=tile_dim1, [207:160]=tensor_dim0_stride
__device__ __forceinline__ void tdm_load_2d(unsigned lds_off, const void* gp,
                                            unsigned tensor_d0, unsigned tensor_d1,
                                            unsigned tile_d0, unsigned tile_d1,
                                            unsigned stride0_elems) {
    unsigned long long ga = (unsigned long long)gp;
    v4u g0;
    g0[0] = 1u;                                            // count=1, user mode
    g0[1] = lds_off;                                       // LDS byte address
    g0[2] = (unsigned)(ga & 0xffffffffu);                  // global_addr[31:0]
    g0[3] = (unsigned)((ga >> 32) & 0x01ffffffu) | (2u << 30);  // [56:32] | type=2
    v8u g1;
    g1[0] = (1u << 16);                                    // workgroup_mask=0, data_size=1 (2B)
    g1[1] = (tensor_d0 & 0xffffu) << 16;                   // tensor_dim0[15:0] -> bits 63:48
    g1[2] = ((tensor_d0 >> 16) & 0xffffu) | ((tensor_d1 & 0xffffu) << 16);
    g1[3] = ((tensor_d1 >> 16) & 0xffffu) | ((tile_d0 & 0xffffu) << 16);
    g1[4] = (tile_d1 & 0xffffu);                           // tile_dim1 | tile_dim2(0)<<16
    g1[5] = stride0_elems;                                 // tensor_dim0_stride[31:0]
    g1[6] = 0u;                                            // stride[47:32] | dim1_stride lo
    g1[7] = 0u;
    asm volatile("tensor_load_to_lds %0, %1" :: "s"(g0), "s"(g1) : "memory");
}

// Store one 16x16 f32 accumulator per documented C/D layout:
// c[r] -> row = mb + r + 8*hi, col = nb + l
template<int MODE>
__device__ __forceinline__ void store_acc(v8f acc, int mb, int nb, int l, int hi,
                                          int N, float scale, int z,
                                          const float* __restrict__ bias,
                                          const float* __restrict__ resid,
                                          float* __restrict__ outF,
                                          bf16* __restrict__ outB) {
    const int col = nb + l;
#pragma unroll
    for (int r = 0; r < 8; r++) {
        const int row = mb + r + 8 * hi;
        float v = acc[r];
        if constexpr (MODE == MODE_SCORES) {
            // attn logits: 512MB streaming tensor -> nontemporal, spare L2
            __builtin_nontemporal_store(v * scale,
                &outF[(long long)z * SEQ * SEQ + (long long)row * N + col]);
        } else if constexpr (MODE == MODE_QK) {
            v += bias[col];
            const int b = row >> 11, s = row & (SEQ - 1);
            const int h = col >> 6,  dk = col & (DK - 1);
            outB[(((long long)b * N_HEADS + h) * SEQ + s) * DK + dk] = (bf16)v;
        } else if constexpr (MODE == MODE_V) {
            v += bias[col];
            const int b = row >> 11, s = row & (SEQ - 1);
            const int h = col >> 6,  dk = col & (DK - 1);
            outB[(((long long)b * N_HEADS + h) * DK + dk) * SEQ + s] = (bf16)v;
        } else if constexpr (MODE == MODE_RELU) {
            v = fmaxf(v + bias[col], 0.0f);
            outB[(long long)row * N + col] = (bf16)v;
        } else if constexpr (MODE == MODE_RESID) {
            v += bias[col] + resid[(long long)row * N + col];
            outF[(long long)row * N + col] = v;
        } else if constexpr (MODE == MODE_CTX) {
            const int b = z >> 4, h = z & (N_HEADS - 1);
            outB[((long long)(b * SEQ + row)) * D_MODEL + h * DK + col] = (bf16)v;
        }
    }
}

// Generic bf16 WMMA GEMM with TDM double-buffered LDS staging.
// C[M,N] = A[M,K] * Bt[N,K]^T (+ epilogue per MODE).
// Block: 256 threads = 8 waves (4 M x 2 N) -> 128 x 64 tile; wave tile 32x32.
// Per K-step (32): wave0 issues 2 TDM tile loads (A: 128x32, B: 64x32) into the
// next LDS buffer while all waves run 4 WMMAs on the current buffer.
template<int MODE>
__global__ __launch_bounds__(256)
void gemm_wmma_k(const bf16* __restrict__ A, const bf16* __restrict__ Bt,
                 const float* __restrict__ bias, const float* __restrict__ resid,
                 float* __restrict__ outF, bf16* __restrict__ outB,
                 int M, int N, int K, float scale,
                 long long strideAz, long long strideBz) {
    __shared__ bf16 sA[2][128 * 32];   // 16 KB
    __shared__ bf16 sB[2][64 * 32];    //  8 KB

    const int z = blockIdx.z;
    const bf16* Az = A + (long long)z * strideAz;
    const bf16* Bz = Bt + (long long)z * strideBz;

    const int wid  = threadIdx.x >> 5;
    const int lane = threadIdx.x & 31;
    const int l  = lane & 15;
    const int hi = lane >> 4;

    const int mBase = blockIdx.x * 128;
    const int nBase = blockIdx.y * 64;

    const bf16* aTile = Az + (long long)mBase * K;   // 128 rows, stride K
    const bf16* bTile = Bz + (long long)nBase * K;   // 64 rows,  stride K

    // LDS fragment row pointers (tile rows packed [row][32])
    const bf16* ar0base = &sA[0][((wid >> 1) * 32 + l) * 32];
    const bf16* br0base = &sB[0][((wid & 1) * 32 + l) * 32];
    const int   bufStrideA = 128 * 32;
    const int   bufStrideB = 64 * 32;

    v8f acc00 = {}, acc01 = {}, acc10 = {}, acc11 = {};

    if (wid == 0) {   // stage first K-slab (TDM, tracked by TENSORcnt)
        tdm_load_2d((unsigned)(unsigned long long)&sA[0][0], aTile,
                    (unsigned)K, 1u << 20, 32u, 128u, (unsigned)K);
        tdm_load_2d((unsigned)(unsigned long long)&sB[0][0], bTile,
                    (unsigned)K, 1u << 20, 32u, 64u, (unsigned)K);
    }

    for (int kb = 0; kb < K; kb += 32) {
        const int cur = (kb >> 5) & 1;
        if (wid == 0) __builtin_amdgcn_s_wait_tensorcnt(0);
        __syncthreads();                       // staged data visible to all waves
        if (wid == 0 && kb + 32 < K) {         // overlap next slab with WMMA
            const int nxt = cur ^ 1;
            tdm_load_2d((unsigned)(unsigned long long)&sA[nxt][0], aTile + kb + 32,
                        (unsigned)K, 1u << 20, 32u, 128u, (unsigned)K);
            tdm_load_2d((unsigned)(unsigned long long)&sB[nxt][0], bTile + kb + 32,
                        (unsigned)K, 1u << 20, 32u, 64u, (unsigned)K);
        }

        const bf16* ar0 = ar0base + cur * bufStrideA;
        const bf16* br0 = br0base + cur * bufStrideB;
        v16bf a0 = load_frag(ar0,            0, hi);
        v16bf a1 = load_frag(ar0 + 16 * 32,  0, hi);
        v16bf b0 = load_frag(br0,            0, hi);
        v16bf b1 = load_frag(br0 + 16 * 32,  0, hi);
        acc00 = wmma_bf16(a0, b0, acc00);
        acc01 = wmma_bf16(a0, b1, acc01);
        acc10 = wmma_bf16(a1, b0, acc10);
        acc11 = wmma_bf16(a1, b1, acc11);
        __syncthreads();                       // readers done before buffer reuse
    }

    const int m0 = mBase + (wid >> 1) * 32;
    const int n0 = nBase + (wid & 1) * 32;
    store_acc<MODE>(acc00, m0,      n0,      l, hi, N, scale, z, bias, resid, outF, outB);
    store_acc<MODE>(acc01, m0,      n0 + 16, l, hi, N, scale, z, bias, resid, outF, outB);
    store_acc<MODE>(acc10, m0 + 16, n0,      l, hi, N, scale, z, bias, resid, outF, outB);
    store_acc<MODE>(acc11, m0 + 16, n0 + 16, l, hi, N, scale, z, bias, resid, outF, outB);
}

// ctx = attn(f32, cvt on the fly, nontemporal) @ V  per (b,h). M=SEQ,N=DK,K=SEQ.
__global__ __launch_bounds__(256)
void ctx_gemm_k(const float* __restrict__ attn, const bf16* __restrict__ Vt,
                bf16* __restrict__ ctxb) {
    const int z = blockIdx.z;                       // b*16 + h
    const float* Az = attn + (long long)z * SEQ * SEQ;
    const bf16*  Bz = Vt   + (long long)z * DK * SEQ;

    const int wid  = threadIdx.x >> 5;
    const int lane = threadIdx.x & 31;
    const int l  = lane & 15;
    const int hi = lane >> 4;

    const int m0 = blockIdx.x * 128 + (wid >> 1) * 32;
    const int n0 = (wid & 1) * 32;                  // N = 64 covered by one block

    const float* a0p = Az + (long long)(m0 + l)      * SEQ;
    const float* a1p = Az + (long long)(m0 + 16 + l) * SEQ;
    const bf16*  b0p = Bz + (long long)(n0 + l)      * SEQ;
    const bf16*  b1p = Bz + (long long)(n0 + 16 + l) * SEQ;

    v8f acc00 = {}, acc01 = {}, acc10 = {}, acc11 = {};

    for (int kb = 0; kb < SEQ; kb += 32) {
        if (kb + 32 < SEQ) {   // global_prefetch_b8 of next V slab
            __builtin_prefetch(b0p + kb + 32, 0, 1);
        }
        v16bf a0 = load_frag_f32(a0p, kb, hi);
        v16bf a1 = load_frag_f32(a1p, kb, hi);
        v16bf b0 = load_frag(b0p, kb, hi);
        v16bf b1 = load_frag(b1p, kb, hi);
        acc00 = wmma_bf16(a0, b0, acc00);
        acc01 = wmma_bf16(a0, b1, acc01);
        acc10 = wmma_bf16(a1, b0, acc10);
        acc11 = wmma_bf16(a1, b1, acc11);
    }

    store_acc<MODE_CTX>(acc00, m0,      n0,      l, hi, DK, 1.f, z, nullptr, nullptr, nullptr, ctxb);
    store_acc<MODE_CTX>(acc01, m0,      n0 + 16, l, hi, DK, 1.f, z, nullptr, nullptr, nullptr, ctxb);
    store_acc<MODE_CTX>(acc10, m0 + 16, n0,      l, hi, DK, 1.f, z, nullptr, nullptr, nullptr, ctxb);
    store_acc<MODE_CTX>(acc11, m0 + 16, n0 + 16, l, hi, DK, 1.f, z, nullptr, nullptr, nullptr, ctxb);
}

// ---- elementwise / reduction kernels ------------------------------------

__global__ __launch_bounds__(256)
void cvt_f32_bf16_k(const float* __restrict__ in, bf16* __restrict__ out, long long n) {
    long long i = (long long)blockIdx.x * 256 + threadIdx.x;
    if (i < n) out[i] = (bf16)in[i];
}

// Wt[n][k] = (bf16) W[k][n]   (W is [K,N] row-major)
__global__ __launch_bounds__(256)
void transpose_cvt_k(const float* __restrict__ W, bf16* __restrict__ Wt, int K, int N) {
    long long i = (long long)blockIdx.x * 256 + threadIdx.x;
    if (i >= (long long)K * N) return;
    int n = (int)(i % N);
    int k = (int)(i / N);
    Wt[(long long)n * K + k] = (bf16)W[i];
}

// In-place softmax over rows of width SEQ. One block (256 thr) per row.
// Entire attn tensor is streaming (512MB) -> nontemporal loads/stores.
__global__ __launch_bounds__(256)
void softmax_rows_k(float* __restrict__ attn) {
    float* row = attn + (long long)blockIdx.x * SEQ;
    const int tid = threadIdx.x;
    __shared__ float red[256];

    float v[SEQ / 256];
    float m = -INFINITY;
#pragma unroll
    for (int i = 0; i < SEQ / 256; i++) {
        v[i] = __builtin_nontemporal_load(&row[tid + 256 * i]);
        m = fmaxf(m, v[i]);
    }
    red[tid] = m; __syncthreads();
    for (int s = 128; s > 0; s >>= 1) { if (tid < s) red[tid] = fmaxf(red[tid], red[tid + s]); __syncthreads(); }
    m = red[0]; __syncthreads();

    float sum = 0.f;
#pragma unroll
    for (int i = 0; i < SEQ / 256; i++) { v[i] = __expf(v[i] - m); sum += v[i]; }
    red[tid] = sum; __syncthreads();
    for (int s = 128; s > 0; s >>= 1) { if (tid < s) red[tid] += red[tid + s]; __syncthreads(); }
    const float inv = 1.0f / red[0];
#pragma unroll
    for (int i = 0; i < SEQ / 256; i++)
        __builtin_nontemporal_store(v[i] * inv, &row[tid + 256 * i]);
}

// LayerNorm per reference: c = y-mean; std = sqrt(sum(c^2)/(D-1)); alpha*c/(std+eps)+beta
__global__ __launch_bounds__(256)
void layernorm_k(const float* __restrict__ y, const float* __restrict__ alpha,
                 const float* __restrict__ beta, float* __restrict__ outF,
                 bf16* __restrict__ outB) {
    const int tid = threadIdx.x;
    const float* row = y + (long long)blockIdx.x * D_MODEL;
    __shared__ float red[256];

    float v[D_MODEL / 256];
    float s = 0.f;
#pragma unroll
    for (int i = 0; i < D_MODEL / 256; i++) { v[i] = row[tid + 256 * i]; s += v[i]; }
    red[tid] = s; __syncthreads();
    for (int k = 128; k > 0; k >>= 1) { if (tid < k) red[tid] += red[tid + k]; __syncthreads(); }
    const float mean = red[0] * (1.0f / D_MODEL); __syncthreads();

    float ss = 0.f;
#pragma unroll
    for (int i = 0; i < D_MODEL / 256; i++) { v[i] -= mean; ss += v[i] * v[i]; }
    red[tid] = ss; __syncthreads();
    for (int k = 128; k > 0; k >>= 1) { if (tid < k) red[tid] += red[tid + k]; __syncthreads(); }
    const float stdv = sqrtf(red[0] * (1.0f / (D_MODEL - 1)));
    const float rinv = 1.0f / (stdv + LN_EPS);

#pragma unroll
    for (int i = 0; i < D_MODEL / 256; i++) {
        const int c = tid + 256 * i;
        const float o = alpha[c] * v[i] * rinv + beta[c];
        outF[(long long)blockIdx.x * D_MODEL + c] = o;
        if (outB) outB[(long long)blockIdx.x * D_MODEL + c] = (bf16)o;
    }
}

// ---------------------------------------------------------------------------

extern "C" void kernel_launch(void* const* d_in, const int* in_sizes, int n_in,
                              void* d_out, int out_size, void* d_ws, size_t ws_size,
                              hipStream_t stream) {
    (void)in_sizes; (void)n_in; (void)out_size; (void)ws_size;

    const float* x  = (const float*)d_in[0];
    const float* Wq = (const float*)d_in[1];  const float* bq = (const float*)d_in[2];
    const float* Wk = (const float*)d_in[3];  const float* bk = (const float*)d_in[4];
    const float* Wv = (const float*)d_in[5];  const float* bv = (const float*)d_in[6];
    const float* Wo = (const float*)d_in[7];  const float* bo = (const float*)d_in[8];
    const float* W1 = (const float*)d_in[9];  const float* b1 = (const float*)d_in[10];
    const float* W2 = (const float*)d_in[11]; const float* b2 = (const float*)d_in[12];
    const float* alpha1 = (const float*)d_in[13]; const float* beta1 = (const float*)d_in[14];
    const float* alpha2 = (const float*)d_in[15]; const float* beta2 = (const float*)d_in[16];

    float* enc_out = (float*)d_out;                                  // [NTOK, D_MODEL]
    float* attn    = enc_out + (size_t)NTOK * D_MODEL;               // [B*H, SEQ, SEQ]

    // ---- carve workspace ----
    char* ws = (char*)d_ws; size_t off = 0;
    auto carve = [&](size_t bytes) -> void* {
        off = (off + 255) & ~(size_t)255;
        void* p = ws + off; off += bytes; return p;
    };
    bf16* xb   = (bf16*)carve((size_t)NTOK * D_MODEL * 2);
    bf16* WqT  = (bf16*)carve((size_t)D_MODEL * D_MODEL * 2);
    bf16* WkT  = (bf16*)carve((size_t)D_MODEL * D_MODEL * 2);
    bf16* WvT  = (bf16*)carve((size_t)D_MODEL * D_MODEL * 2);
    bf16* WoT  = (bf16*)carve((size_t)D_MODEL * D_MODEL * 2);
    bf16* W1T  = (bf16*)carve((size_t)D_FF * D_MODEL * 2);           // [4096][1024]
    bf16* W2T  = (bf16*)carve((size_t)D_MODEL * D_FF * 2);           // [1024][4096]
    bf16* Qb   = (bf16*)carve((size_t)NTOK * D_MODEL * 2);           // [B,H,S,dk]
    bf16* Kb   = (bf16*)carve((size_t)NTOK * D_MODEL * 2);           // [B,H,S,dk]
    bf16* Vtb  = (bf16*)carve((size_t)NTOK * D_MODEL * 2);           // [B,H,dk,S]
    bf16* ctxb = (bf16*)carve((size_t)NTOK * D_MODEL * 2);           // [NTOK, D]
    float* y1f = (float*)carve((size_t)NTOK * D_MODEL * 4);          // x + ctx@Wo
    float* aof = (float*)carve((size_t)NTOK * D_MODEL * 4);          // LN1 out f32
    bf16*  aob = (bf16*)carve((size_t)NTOK * D_MODEL * 2);           // LN1 out bf16
    bf16*  ff1 = (bf16*)carve((size_t)NTOK * D_FF * 2);              // relu(ao@W1)
    float* y2f = (float*)carve((size_t)NTOK * D_MODEL * 4);          // ao + ff

    const dim3 blk(256);

    // 1) precision conversion + weight transposes
    cvt_f32_bf16_k<<<(NTOK * D_MODEL) / 256, blk, 0, stream>>>(x, xb, (long long)NTOK * D_MODEL);
    transpose_cvt_k<<<(D_MODEL * D_MODEL) / 256, blk, 0, stream>>>(Wq, WqT, D_MODEL, D_MODEL);
    transpose_cvt_k<<<(D_MODEL * D_MODEL) / 256, blk, 0, stream>>>(Wk, WkT, D_MODEL, D_MODEL);
    transpose_cvt_k<<<(D_MODEL * D_MODEL) / 256, blk, 0, stream>>>(Wv, WvT, D_MODEL, D_MODEL);
    transpose_cvt_k<<<(D_MODEL * D_MODEL) / 256, blk, 0, stream>>>(Wo, WoT, D_MODEL, D_MODEL);
    transpose_cvt_k<<<(D_MODEL * D_FF) / 256, blk, 0, stream>>>(W1, W1T, D_MODEL, D_FF);
    transpose_cvt_k<<<(D_FF * D_MODEL) / 256, blk, 0, stream>>>(W2, W2T, D_FF, D_MODEL);

    // 2) Q/K/V projections (M=4096, N=1024, K=1024)
    {
        dim3 g(NTOK / 128, D_MODEL / 64, 1);
        gemm_wmma_k<MODE_QK><<<g, blk, 0, stream>>>(xb, WqT, bq, nullptr, nullptr, Qb,
                                                    NTOK, D_MODEL, D_MODEL, 1.f, 0, 0);
        gemm_wmma_k<MODE_QK><<<g, blk, 0, stream>>>(xb, WkT, bk, nullptr, nullptr, Kb,
                                                    NTOK, D_MODEL, D_MODEL, 1.f, 0, 0);
        gemm_wmma_k<MODE_V><<<g, blk, 0, stream>>>(xb, WvT, bv, nullptr, nullptr, Vtb,
                                                   NTOK, D_MODEL, D_MODEL, 1.f, 0, 0);
    }

    // 3) scores = Q K^T / sqrt(dk)  (per head: M=N=2048, K=64)
    {
        dim3 g(SEQ / 128, SEQ / 64, BATCH * N_HEADS);
        gemm_wmma_k<MODE_SCORES><<<g, blk, 0, stream>>>(Qb, Kb, nullptr, nullptr, attn, nullptr,
                                                        SEQ, SEQ, DK, 0.125f,
                                                        (long long)SEQ * DK, (long long)SEQ * DK);
    }

    // 4) softmax over last dim
    softmax_rows_k<<<BATCH * N_HEADS * SEQ, blk, 0, stream>>>(attn);

    // 5) ctx = attn @ V  (per head: M=2048, N=64, K=2048)
    {
        dim3 g(SEQ / 128, 1, BATCH * N_HEADS);
        ctx_gemm_k<<<g, blk, 0, stream>>>(attn, Vtb, ctxb);
    }

    // 6) y1 = x + ctx @ Wo + bo ; attn_out = LN1(y1)
    {
        dim3 g(NTOK / 128, D_MODEL / 64, 1);
        gemm_wmma_k<MODE_RESID><<<g, blk, 0, stream>>>(ctxb, WoT, bo, x, y1f, nullptr,
                                                       NTOK, D_MODEL, D_MODEL, 1.f, 0, 0);
    }
    layernorm_k<<<NTOK, blk, 0, stream>>>(y1f, alpha1, beta1, aof, aob);

    // 7) ff1 = relu(attn_out @ W1 + b1)   (M=4096, N=4096, K=1024)
    {
        dim3 g(NTOK / 128, D_FF / 64, 1);
        gemm_wmma_k<MODE_RELU><<<g, blk, 0, stream>>>(aob, W1T, b1, nullptr, nullptr, ff1,
                                                      NTOK, D_FF, D_MODEL, 1.f, 0, 0);
    }

    // 8) y2 = attn_out + ff1 @ W2 + b2    (M=4096, N=1024, K=4096)
    {
        dim3 g(NTOK / 128, D_MODEL / 64, 1);
        gemm_wmma_k<MODE_RESID><<<g, blk, 0, stream>>>(ff1, W2T, b2, aof, y2f, nullptr,
                                                       NTOK, D_MODEL, D_FF, 1.f, 0, 0);
    }

    // 9) enc_out = LN2(y2)
    layernorm_k<<<NTOK, blk, 0, stream>>>(y2f, alpha2, beta2, enc_out, nullptr);
}
